// NeighborAttentionV2_13881334301001
// MI455X (gfx1250) — matrix operand order repaired and checked by reading, not compiled, not run on hardware
//
#include <hip/hip_runtime.h>
#include <hip/hip_bf16.h>
#include <math.h>

// ---------------------------------------------------------------------------
// Types for CDNA5 WMMA (wave32, 16x16x32 bf16 -> f32)
// ---------------------------------------------------------------------------
typedef __bf16 bhalf;
typedef __attribute__((ext_vector_type(16))) __bf16 v16bf;
typedef __attribute__((ext_vector_type(8)))  float  v8f;

#define NUM_HIDDEN 128
#define NUM_IN     256
#define NUM_HEADS  4
#define EM         64      // edges per block in the fused edge-MLP kernel
#define XLD        392     // LDS row stride (bf16 elems): %8==0 (b128 DS), 196 dwords -> 4-bank row skew

// float <-> order-preserving unsigned encoding (for atomicMax on floats)
__device__ __forceinline__ unsigned encf(float f) {
  unsigned b = __float_as_uint(f);
  return (b & 0x80000000u) ? ~b : (b | 0x80000000u);
}
__device__ __forceinline__ float decf(unsigned u) {
  unsigned b = (u & 0x80000000u) ? (u & 0x7FFFFFFFu) : ~u;
  return __uint_as_float(b);
}
#define ENC_NEG_INF 0x007FFFFFu   // encf(-inf)

__device__ __forceinline__ float gelu_exact(float x) {
  return 0.5f * x * (1.0f + erff(x * 0.70710678118654752f));
}

// ---------------------------------------------------------------------------
// WMMA fragment gathers (A/B: same contiguous k order, not the ISA table order; C/D per ISA §7.12.2)
// A (16x32 bf16, MxK): lane%16 = M; lanes 0-15 hold K=0..15 (2/VGPR), lanes 16-31 K=16..31
// B (32x16 bf16, KxN): lane%16 = N; lanes 0-15 hold K=0..15 (2/VGPR), lanes 16-31 K=16..31
// C/D (16x16 f32):     lane%16 = N; VGPR r: M = r + 8*(lane>>4)
// ---------------------------------------------------------------------------
__device__ __forceinline__ v16bf ldsA(const bhalf* base, int ldx, int lane, int kbase) {
  const int m = lane & 15, half = lane >> 4;
  const bhalf* p = base + m * ldx + kbase + 16 * half;
  v16bf a;
#pragma unroll
  for (int e = 0; e < 16; ++e) {
    const int ko = e; // p[0..15]
    a[e] = p[ko];
  }
  return a;
}

template <bool FULL>
__device__ __forceinline__ v16bf gblB(const float* __restrict__ W, int ldw,
                                      int n0, int kbase, int lane, int nvalid) {
  const int n = n0 + (lane & 15);
  const int half = lane >> 4;
  v16bf b;
  if (FULL || n < nvalid) {
    const float* p = W + (long long)n * ldw + kbase + 16 * half; // 16 consecutive floats
#pragma unroll
    for (int e = 0; e < 16; ++e) b[e] = (bhalf)p[e];
  } else {
#pragma unroll
    for (int e = 0; e < 16; ++e) b[e] = (bhalf)0.0f;
  }
  return b;
}

// One wave: 16-wide output column tile n0, all 4 M-tiles (64 rows), K-loop of x32 steps.
// FULL=true: every lane's column is in-range (no EXEC divergence in the hot loop).
template <bool FULL>
__device__ __forceinline__ void gemm_col(const bhalf* inbase, int ldx, int K,
                                         const float* __restrict__ W,
                                         const float* __restrict__ bias,
                                         int n0, int nvalid, int lane, v8f acc[4]) {
  const int n = n0 + (lane & 15);
  const float b0 = (bias != nullptr && (FULL || n < nvalid)) ? bias[n] : 0.0f;
#pragma unroll
  for (int mt = 0; mt < 4; ++mt)
#pragma unroll
    for (int r = 0; r < 8; ++r) acc[mt][r] = b0;

  for (int k = 0; k < K; k += 32) {
    const v16bf b = gblB<FULL>(W, K, n0, k, lane, nvalid); // weights are L2-resident (0.8 MB)
    v16bf a[4];
#pragma unroll
    for (int mt = 0; mt < 4; ++mt)            // issue all 8 ds_load_b128 up front
      a[mt] = ldsA(inbase + mt * 16 * ldx, ldx, lane, k);
#pragma unroll
    for (int mt = 0; mt < 4; ++mt)
      acc[mt] = __builtin_amdgcn_wmma_f32_16x16x32_bf16(
          false, a[mt], false, b, (short)0, acc[mt], false, false);
  }
}

__device__ __forceinline__ void store_lds_tiles(bhalf* out, int ldx, int lane,
                                                const v8f acc[4], bool act) {
  const int n = lane & 15, half = lane >> 4;
#pragma unroll
  for (int mt = 0; mt < 4; ++mt)
#pragma unroll
    for (int r = 0; r < 8; ++r) {
      float v = acc[mt][r];
      if (act) v = gelu_exact(v);
      out[(mt * 16 + r + 8 * half) * ldx + n] = (bhalf)v;
    }
}

// ---------------------------------------------------------------------------
// Kernel 0: workspace init (graph-replay safe: re-run every launch)
// ---------------------------------------------------------------------------
__global__ void init_ws_kernel(unsigned* __restrict__ segmax, float* __restrict__ segsum,
                               float* __restrict__ agg, int N) {
  const long long i = (long long)blockIdx.x * blockDim.x + threadIdx.x;
  if (i < (long long)N * NUM_HIDDEN) agg[i] = 0.0f;
  if (i < (long long)N * NUM_HEADS) { segsum[i] = 0.0f; segmax[i] = ENC_NEG_INF; }
}

// ---------------------------------------------------------------------------
// Kernel 1: fused 6-layer edge MLP (all GEMMs via v_wmma_f32_16x16x32_bf16)
// One block = 64 edges, 8 waves. Single 50 KB LDS activation buffer,
// barrier-phased in-place layer pipeline.
// ---------------------------------------------------------------------------
__global__ __launch_bounds__(256) void edge_mlp_kernel(
    const float* __restrict__ hV, const float* __restrict__ hE, const int* __restrict__ center,
    const float* __restrict__ Wm1, const float* __restrict__ bm1,
    const float* __restrict__ Wm2, const float* __restrict__ bm2,
    const float* __restrict__ Wm3, const float* __restrict__ bm3,
    const float* __restrict__ Wv1, const float* __restrict__ bv1,
    const float* __restrict__ Wv2, const float* __restrict__ bv2,
    const float* __restrict__ Wv3, const float* __restrict__ bv3,
    float* __restrict__ logits, float* __restrict__ Vout,
    unsigned* __restrict__ segmax, int E) {
  __shared__ bhalf xs[EM * XLD];   // 64 x 384 activations (padded)
  __shared__ int ce[EM];

  const int tid = threadIdx.x;
  const int lane = tid & 31;
  const int w = tid >> 5;                       // wave id 0..7
  const long long e0 = (long long)blockIdx.x * EM;

  if (tid < EM) {
    const long long e = e0 + tid;
    ce[tid] = (e < E) ? center[e] : 0;
  }
  __syncthreads();

  // ---- stage x = concat(hV[center], hE) as bf16, float4 loads ----
  for (int q = tid; q < EM * 96; q += 256) {    // 96 float4 per 384-col row
    const int row = q / 96;
    const int c4 = (q % 96) * 4;
    const long long e = e0 + row;
    float4 v = make_float4(0.f, 0.f, 0.f, 0.f);
    if (e < E) {
      if (c4 < NUM_HIDDEN) v = *(const float4*)(hV + (long long)ce[row] * NUM_HIDDEN + c4);
      else                 v = *(const float4*)(hE + e * NUM_IN + (c4 - NUM_HIDDEN));
    }
    bhalf* d = xs + row * XLD + c4;
    d[0] = (bhalf)v.x; d[1] = (bhalf)v.y; d[2] = (bhalf)v.z; d[3] = (bhalf)v.w;
  }
  __syncthreads();

  // ---- phase 1: m1 (384->256 gelu) + v1 (256->128 gelu); x read by both ----
  v8f am1a[4], am1b[4], av1[4];
  gemm_col<true>(xs,        XLD, 384, Wm1, bm1, (2 * w) * 16,     256, lane, am1a);
  gemm_col<true>(xs,        XLD, 384, Wm1, bm1, (2 * w + 1) * 16, 256, lane, am1b);
  gemm_col<true>(xs + 128,  XLD, 256, Wv1, bv1, w * 16,           128, lane, av1);
  __syncthreads();                              // all reads of x done
  store_lds_tiles(xs + (2 * w) * 16,     XLD, lane, am1a, true);   // cols 0..255
  store_lds_tiles(xs + (2 * w + 1) * 16, XLD, lane, am1b, true);
  store_lds_tiles(xs + 256 + w * 16,     XLD, lane, av1,  true);   // cols 256..383
  __syncthreads();

  // ---- phase 2: m2 (256->128 gelu) + v2 (128->128 gelu) ----
  v8f am2[4], av2[4];
  gemm_col<true>(xs,       XLD, 256, Wm2, bm2, w * 16, 128, lane, am2);
  gemm_col<true>(xs + 256, XLD, 128, Wv2, bv2, w * 16, 128, lane, av2);
  __syncthreads();
  store_lds_tiles(xs + w * 16,       XLD, lane, am2, true);        // cols 0..127
  store_lds_tiles(xs + 128 + w * 16, XLD, lane, av2, true);        // cols 128..255
  __syncthreads();

  // ---- phase 3: v3 (128->128, no act) -> global V ----
  v8f av3[4];
  gemm_col<true>(xs + 128, XLD, 128, Wv3, bv3, w * 16, 128, lane, av3);
  {
    const int n = lane & 15, half = lane >> 4;
#pragma unroll
    for (int mt = 0; mt < 4; ++mt)
#pragma unroll
      for (int r = 0; r < 8; ++r) {
        const long long e = e0 + mt * 16 + r + 8 * half;
        if (e < E) Vout[e * NUM_HIDDEN + w * 16 + n] = av3[mt][r];
      }
  }

  // ---- phase 3b: m3 (128->4) -> scaled logits + segment atomic-max (wave 0) ----
  if (w == 0) {
    v8f am3[4];
    gemm_col<false>(xs, XLD, 128, Wm3, bm3, 0, NUM_HEADS, lane, am3); // n>=4 zero-padded
    const int n = lane & 15, half = lane >> 4;
    const float scale = 0.17677669529663687f;                      // 1/sqrt(d), d=32
    if (n < NUM_HEADS) {
#pragma unroll
      for (int mt = 0; mt < 4; ++mt)
#pragma unroll
        for (int r = 0; r < 8; ++r) {
          const int row = mt * 16 + r + 8 * half;
          const long long e = e0 + row;
          if (e < E) {
            const float lg = am3[mt][r] * scale;
            logits[e * NUM_HEADS + n] = lg;
            atomicMax(&segmax[(long long)ce[row] * NUM_HEADS + n], encf(lg));
          }
        }
    }
  }
}

// ---------------------------------------------------------------------------
// Kernel 2: ex = exp(logit - segmax[center]); segment-sum via atomics (in-place)
// ---------------------------------------------------------------------------
__global__ void softmax_sum_kernel(const int* __restrict__ center, float* __restrict__ lg,
                                   const unsigned* __restrict__ segmax,
                                   float* __restrict__ segsum, int E) {
  const int i = blockIdx.x * blockDim.x + threadIdx.x;
  if (i >= E * NUM_HEADS) return;
  const int e = i >> 2, h = i & 3;
  const int c = center[e];
  const float m = decf(segmax[c * NUM_HEADS + h]);
  const float ex = expf(lg[i] - m);
  lg[i] = ex;                                   // logits buffer now holds ex
  atomicAdd(&segsum[c * NUM_HEADS + h], ex);
}

// ---------------------------------------------------------------------------
// Kernel 3: agg[c] += (ex/segsum[c]) * V[e]   (128 lanes per edge, coalesced V)
// ---------------------------------------------------------------------------
__global__ void scatter_kernel(const int* __restrict__ center, const float* __restrict__ ex,
                               const float* __restrict__ segsum, const float* __restrict__ V,
                               float* __restrict__ agg, int E) {
  const long long i = (long long)blockIdx.x * blockDim.x + threadIdx.x;
  if (i >= (long long)E * NUM_HIDDEN) return;
  const int e = (int)(i >> 7);
  const int j = (int)(i & 127);
  const int h = j >> 5;                         // head = j / 32
  const int c = center[e];
  const float a = ex[e * NUM_HEADS + h] / segsum[c * NUM_HEADS + h];
  atomicAdd(&agg[(long long)c * NUM_HIDDEN + j], a * V[i]);
}

// ---------------------------------------------------------------------------
// Kernel 4: h_out = agg @ Wo.T  (WMMA; one 16-row tile per block, 8 N-tiles/waves)
// ---------------------------------------------------------------------------
__global__ __launch_bounds__(256) void out_gemm_kernel(const float* __restrict__ agg,
                                                       const float* __restrict__ Wo,
                                                       float* __restrict__ out, int N) {
  __shared__ bhalf as[16 * 136];                // 136 bf16 stride: %8==0, 68-dword row skew
  const int tid = threadIdx.x, lane = tid & 31, w = tid >> 5;
  const long long m0 = (long long)blockIdx.x * 16;

  for (int q = tid; q < 16 * 32; q += 256) {    // 32 float4 per 128-col row
    const int row = q / 32;
    const int c4 = (q % 32) * 4;
    const long long m = m0 + row;
    float4 v = (m < N) ? *(const float4*)(agg + m * NUM_HIDDEN + c4)
                       : make_float4(0.f, 0.f, 0.f, 0.f);
    bhalf* d = as + row * 136 + c4;
    d[0] = (bhalf)v.x; d[1] = (bhalf)v.y; d[2] = (bhalf)v.z; d[3] = (bhalf)v.w;
  }
  __syncthreads();

  v8f acc;
#pragma unroll
  for (int r = 0; r < 8; ++r) acc[r] = 0.0f;
  for (int k = 0; k < 128; k += 32) {
    const v16bf b = gblB<true>(Wo, 128, w * 16, k, lane, 128);
    const v16bf a = ldsA(as, 136, lane, k);
    acc = __builtin_amdgcn_wmma_f32_16x16x32_bf16(false, a, false, b, (short)0, acc,
                                                  false, false);
  }
  const int n = lane & 15, half = lane >> 4;
#pragma unroll
  for (int r = 0; r < 8; ++r) {
    const long long m = m0 + r + 8 * half;
    if (m < N) out[m * NUM_HIDDEN + w * 16 + n] = acc[r];
  }
}

// ---------------------------------------------------------------------------
// Host entry.  Input order (setup_inputs): h_V, h_E, center_id, Wv1, bv1, Wv2,
// bv2, Wv3, bv3, Wm1, bm1, Wm2, bm2, Wm3, bm3, Wo.
// Workspace (fp32): logits/ex E*4 | V E*128 | segmax N*4 (u32) | segsum N*4 | agg N*128
//   => ~450 MB for E=800k, N=50k.
// ---------------------------------------------------------------------------
extern "C" void kernel_launch(void* const* d_in, const int* in_sizes, int n_in,
                              void* d_out, int out_size, void* d_ws, size_t ws_size,
                              hipStream_t stream) {
  const float* hV  = (const float*)d_in[0];
  const float* hE  = (const float*)d_in[1];
  const int* center = (const int*)d_in[2];
  const float* Wv1 = (const float*)d_in[3];  const float* bv1 = (const float*)d_in[4];
  const float* Wv2 = (const float*)d_in[5];  const float* bv2 = (const float*)d_in[6];
  const float* Wv3 = (const float*)d_in[7];  const float* bv3 = (const float*)d_in[8];
  const float* Wm1 = (const float*)d_in[9];  const float* bm1 = (const float*)d_in[10];
  const float* Wm2 = (const float*)d_in[11]; const float* bm2 = (const float*)d_in[12];
  const float* Wm3 = (const float*)d_in[13]; const float* bm3 = (const float*)d_in[14];
  const float* Wo  = (const float*)d_in[15];

  const int N = in_sizes[0] / NUM_HIDDEN;
  const int E = in_sizes[2];

  float* logits    = (float*)d_ws;                               // E*4  (becomes ex)
  float* Vws       = logits + (long long)E * NUM_HEADS;          // E*128
  unsigned* segmax = (unsigned*)(Vws + (long long)E * NUM_HIDDEN); // N*4
  float* segsum    = (float*)(segmax + (long long)N * NUM_HEADS);  // N*4
  float* agg       = segsum + (long long)N * NUM_HEADS;            // N*128
  (void)ws_size; (void)n_in; (void)out_size;

  // 0) init segment state + agg
  {
    const long long tot = (long long)N * NUM_HIDDEN;
    init_ws_kernel<<<(unsigned)((tot + 255) / 256), 256, 0, stream>>>(segmax, segsum, agg, N);
  }
  // 1) fused edge MLP -> logits, V, segment max
  edge_mlp_kernel<<<(unsigned)((E + EM - 1) / EM), 256, 0, stream>>>(
      hV, hE, center, Wm1, bm1, Wm2, bm2, Wm3, bm3,
      Wv1, bv1, Wv2, bv2, Wv3, bv3, logits, Vws, segmax, E);
  // 2) exp + segment sum
  softmax_sum_kernel<<<(unsigned)(((long long)E * NUM_HEADS + 255) / 256), 256, 0, stream>>>(
      center, logits, segmax, segsum, E);
  // 3) attend * V scatter into agg
  scatter_kernel<<<(unsigned)(((long long)E * NUM_HIDDEN + 255) / 256), 256, 0, stream>>>(
      center, logits, segsum, Vws, agg, E);
  // 4) output projection
  out_gemm_kernel<<<(unsigned)((N + 15) / 16), 256, 0, stream>>>(agg, Wo, (float*)d_out, N);
}
